// sim_cell_83700322664663
// MI455X (gfx1250) — compile-verified, matrix operand order checked by reading, not run on hardware
//
#include <hip/hip_runtime.h>

// ---------------------------------------------------------------------------
// Problem constants (from reference): S=16384, I=256, H=512
// ---------------------------------------------------------------------------
constexpr int S_DIM = 16384;
constexpr int I_DIM = 256;
constexpr int H_DIM = 512;

typedef __bf16 bf16_t;
typedef __attribute__((ext_vector_type(16))) __bf16 bf16x16;
typedef __attribute__((ext_vector_type(8)))  float  floatx8;

// ---------------------------------------------------------------------------
// Helpers
// ---------------------------------------------------------------------------
__device__ __forceinline__ unsigned short bf16_bits(float f) {
  unsigned u = __float_as_uint(f);
  u += 0x7FFFu + ((u >> 16) & 1u);  // round-to-nearest-even
  return (unsigned short)(u >> 16);
}

__device__ __forceinline__ bf16_t f32_to_bf16(float f) {
  unsigned short s = bf16_bits(f);
  return __builtin_bit_cast(bf16_t, s);
}

__device__ __forceinline__ float sigmoidf_fast(float x) {
  return 1.0f / (1.0f + __expf(-x));
}

__device__ __forceinline__ float tanhf_fast(float x) {
  float e = __expf(2.0f * x);
  return 1.0f - 2.0f / (e + 1.0f);  // saturates correctly for |x| large
}

__device__ __forceinline__ floatx8 wmma_bf16(bf16x16 a, bf16x16 b, floatx8 c) {
  // (neg_a, A, neg_b, B, c_mod, C, reuse_a, reuse_b)
  return __builtin_amdgcn_wmma_f32_16x16x32_bf16(false, a, false, b,
                                                 (short)0, c, false, false);
}

// CDNA5 async global->LDS copy, 16 bytes per lane, tracked by ASYNCcnt.
// GV addressing mode: LDS address taken from the VDST VGPR, global address from the VADDR VGPRs.
__device__ __forceinline__ void async_copy_b128(unsigned lds_addr,
                                                const void* gaddr) {
  asm volatile("global_load_async_to_lds_b128 %0, %1, off"
               :: "v"(lds_addr), "v"(gaddr)
               : "memory");
}

__device__ __forceinline__ void wait_async0() {
  asm volatile("s_wait_asynccnt 0x0" ::: "memory");
}

// Low 32 bits of a generic pointer to __shared__ data = LDS byte address
// (see ISA section 10.2).
__device__ __forceinline__ unsigned lds_addr_of(const void* p) {
  return (unsigned)(size_t)p;
}

// Load a 16x32 bf16 fragment (A-layout per CDNA5 ISA 7.12.2):
//  lanes 0-15: row M = lane,    K = {0..7, 16..23}
//  lanes16-31: row M = lane-16, K = {8..15, 24..31}
// Same pattern serves the B operand with "row" = output column, reading W in
// its native row-major [N, K] layout (B = W^T).
__device__ __forceinline__ bf16x16 load_frag(const bf16_t* __restrict__ base,
                                             int ld, int row0, int k0, int lane) {
  int r  = row0 + (lane & 15);
  int kh = ((lane >> 4) & 1) << 3;  // 0 or 8
  const bf16_t* p = base + (size_t)r * ld + k0 + kh;
  union { uint4 u[2]; bf16x16 v; } tmp;
  tmp.u[0] = *(const uint4*)(p);        // K = kh .. kh+7
  tmp.u[1] = *(const uint4*)(p + 16);   // K = 16+kh .. 16+kh+7
  return tmp.v;
}

// ---------------------------------------------------------------------------
// fp32 -> bf16 streaming convert (vectorized x4)
// ---------------------------------------------------------------------------
__global__ __launch_bounds__(256) void cvt_f32_bf16_x4(
    const float4* __restrict__ src, uint2* __restrict__ dst, long n4) {
  long i = (long)blockIdx.x * blockDim.x + threadIdx.x;
  long stride = (long)gridDim.x * blockDim.x;
  for (; i < n4; i += stride) {
    float4 v = src[i];
    unsigned lo = (unsigned)bf16_bits(v.x) | ((unsigned)bf16_bits(v.y) << 16);
    unsigned hi = (unsigned)bf16_bits(v.z) | ((unsigned)bf16_bits(v.w) << 16);
    dst[i] = make_uint2(lo, hi);
  }
}

// ---------------------------------------------------------------------------
// Kernel 1: spatial gates
//   spatial_f = sigmoid([hA|hB] @ W_sf^T + b_sf)
//   spatial_i = sigmoid([hA|hB] @ W_si^T + b_si)
//   cbuf = c_s * spatial_f   (fp32, staged in d_out's c_new half)
//   hout = h_s * spatial_i   (bf16, feeds gates GEMM)
// Block: 256 threads (8 waves), tile 32 rows x 64 cols of H.
// A tiles are double-buffered in LDS via async global->LDS copies (ASYNCcnt).
// ---------------------------------------------------------------------------
__global__ __launch_bounds__(256) void spatial_kernel(
    const bf16_t* __restrict__ hA,   // [S, 512] bf16
    const bf16_t* __restrict__ hB,   // [S, 512] bf16
    const bf16_t* __restrict__ Wsf,  // [512, 1024] bf16 row-major
    const bf16_t* __restrict__ Wsi,  // [512, 1024] bf16 row-major
    const float*  __restrict__ b_sf, // [512]
    const float*  __restrict__ b_si, // [512]
    const float*  __restrict__ c_s,  // [S, 512] f32
    const float*  __restrict__ h_s,  // [S, 512] f32
    float*        __restrict__ cbuf, // [S, 512] f32 out (c = c_s * sf)
    bf16_t*       __restrict__ hout) // [S, 512] bf16 out (h = h_s * si)
{
  __shared__ bf16_t As[2][32 * 64];  // 8 KB double-buffered A staging

  const int row0 = blockIdx.x * 32;
  const int col0 = blockIdx.y * 64;
  const int wave = threadIdx.x >> 5;
  const int lane = threadIdx.x & 31;
  const int m_i  = wave >> 2;          // 0..1
  const int n_i  = wave & 3;           // 0..3
  const int wrow = row0 + m_i * 16;
  const int wcol = col0 + n_i * 16;

  floatx8 acc_f = {};
  floatx8 acc_i = {};

  // per-thread 16B slot of the cooperative 32x64 A tile
  const int rr = threadIdx.x >> 3;        // 0..31
  const int cc = (threadIdx.x & 7) << 3;  // 0,8,..,56
  const unsigned ldsSlot[2] = { lds_addr_of(&As[0][rr * 64 + cc]),
                                lds_addr_of(&As[1][rr * 64 + cc]) };

  // chunk ci in [0,16): phase = ci>>3 (hA then hB), kc = (ci&7)*64
  auto gsrc = [&](int ci) -> const void* {
    const bf16_t* Ap = (ci & 8) ? hB : hA;
    int kc = (ci & 7) << 6;
    return (const void*)&Ap[(size_t)(row0 + rr) * 512 + kc + cc];
  };

  constexpr int NCHUNK = 16;
  async_copy_b128(ldsSlot[0], gsrc(0));

  for (int ci = 0; ci < NCHUNK; ++ci) {
    const int cur = ci & 1;
    wait_async0();      // my async fill of buf[cur] landed
    __syncthreads();    // everyone's fills landed; prior reads of buf[1-cur] done
    if (ci + 1 < NCHUNK) async_copy_b128(ldsSlot[1 - cur], gsrc(ci + 1));

    const bf16_t* Abuf = &As[cur][0];
    const int kw = (ci << 6) & 1023;  // K offset inside [512,1024] weights
#pragma unroll
    for (int ks = 0; ks < 64; ks += 32) {
      bf16x16 a  = load_frag(Abuf, 64, m_i * 16, ks, lane);
      bf16x16 bf = load_frag(Wsf, 1024, wcol, kw + ks, lane);
      bf16x16 bi = load_frag(Wsi, 1024, wcol, kw + ks, lane);
      acc_f = wmma_bf16(a, bf, acc_f);
      acc_i = wmma_bf16(a, bi, acc_i);
    }
    if (ci + 1 < NCHUNK) {  // warm L2 with next weight chunk
      __builtin_prefetch(Wsf + (size_t)wcol * 1024 + ((ci + 1) << 6), 0, 1);
      __builtin_prefetch(Wsi + (size_t)wcol * 1024 + ((ci + 1) << 6), 0, 1);
    }
    __syncthreads();    // done reading buf[cur] before it is refilled
  }

  // Epilogue: C/D layout -> lane holds N = lane&15, M = (lane>>4)*8 + i
  const int n     = lane & 15;
  const int mbase = (lane >> 4) * 8;
  const int col   = wcol + n;
  const float bfv = b_sf[col];
  const float biv = b_si[col];
#pragma unroll
  for (int i = 0; i < 8; ++i) {
    const int row  = wrow + mbase + i;
    const size_t idx = (size_t)row * H_DIM + col;
    float sf = sigmoidf_fast(acc_f[i] + bfv);
    float si = sigmoidf_fast(acc_i[i] + biv);
    cbuf[idx] = c_s[idx] * sf;
    hout[idx] = f32_to_bf16(h_s[idx] * si);
  }
}

// ---------------------------------------------------------------------------
// Kernel 2: LSTM gates, fully fused
//   gates = X @ W_ih^T + h @ W_hh^T + (b_ih + b_hh)   -> i,f,g,o stripes
//   c_new = sigmoid(f)*c + sigmoid(i)*tanh(g)
//   h_new = sigmoid(o)*tanh(c_new)
// Each wave owns a 16x16 tile of all FOUR gate stripes: one A fragment feeds
// 4 WMMAs per K-step; gates never touch memory. Async double-buffered A.
// ---------------------------------------------------------------------------
__global__ __launch_bounds__(256) void gates_kernel(
    const bf16_t* __restrict__ Xin,  // [S, 256] bf16
    const bf16_t* __restrict__ Hin,  // [S, 512] bf16 (gated h)
    const bf16_t* __restrict__ Wih,  // [2048, 256] bf16 row-major
    const bf16_t* __restrict__ Whh,  // [2048, 512] bf16 row-major
    const float*  __restrict__ b_ih, // [2048]
    const float*  __restrict__ b_hh, // [2048]
    float*        __restrict__ cbuf, // [S, 512] f32: reads c, writes c_new
    float*        __restrict__ hnew) // [S, 512] f32: writes h_new
{
  __shared__ bf16_t As[2][32 * 64];  // 8 KB double-buffered A staging

  const int row0 = blockIdx.x * 32;
  const int col0 = blockIdx.y * 64;  // within H
  const int wave = threadIdx.x >> 5;
  const int lane = threadIdx.x & 31;
  const int m_i  = wave >> 2;
  const int n_i  = wave & 3;
  const int wrow = row0 + m_i * 16;
  const int wcol = col0 + n_i * 16;  // column within H

  floatx8 acc[4] = {};  // i, f, g, o

  const int rr = threadIdx.x >> 3;
  const int cc = (threadIdx.x & 7) << 3;
  const unsigned ldsSlot[2] = { lds_addr_of(&As[0][rr * 64 + cc]),
                                lds_addr_of(&As[1][rr * 64 + cc]) };

  // chunk ci in [0,12): ci<4 -> Xin (K=256), else Hin (K=512)
  auto gsrc = [&](int ci) -> const void* {
    if (ci < 4)
      return (const void*)&Xin[(size_t)(row0 + rr) * I_DIM + (ci << 6) + cc];
    return (const void*)&Hin[(size_t)(row0 + rr) * H_DIM + ((ci - 4) << 6) + cc];
  };

  constexpr int NCHUNK = 12;
  async_copy_b128(ldsSlot[0], gsrc(0));

  for (int ci = 0; ci < NCHUNK; ++ci) {
    const int cur = ci & 1;
    wait_async0();
    __syncthreads();
    if (ci + 1 < NCHUNK) async_copy_b128(ldsSlot[1 - cur], gsrc(ci + 1));

    const bf16_t* Abuf = &As[cur][0];
    const bf16_t* Wp = (ci < 4) ? Wih : Whh;
    const int ldw    = (ci < 4) ? I_DIM : H_DIM;
    const int kw     = (ci < 4) ? (ci << 6) : ((ci - 4) << 6);
#pragma unroll
    for (int ks = 0; ks < 64; ks += 32) {
      bf16x16 a = load_frag(Abuf, 64, m_i * 16, ks, lane);
#pragma unroll
      for (int g = 0; g < 4; ++g) {
        bf16x16 b = load_frag(Wp, ldw, g * H_DIM + wcol, kw + ks, lane);
        acc[g] = wmma_bf16(a, b, acc[g]);
      }
    }
    __syncthreads();
  }

  // ---- Fused LSTM elementwise epilogue ----
  const int n     = lane & 15;
  const int mbase = (lane >> 4) * 8;
  const int colH  = wcol + n;
  const float bias_i = b_ih[0 * H_DIM + colH] + b_hh[0 * H_DIM + colH];
  const float bias_f = b_ih[1 * H_DIM + colH] + b_hh[1 * H_DIM + colH];
  const float bias_g = b_ih[2 * H_DIM + colH] + b_hh[2 * H_DIM + colH];
  const float bias_o = b_ih[3 * H_DIM + colH] + b_hh[3 * H_DIM + colH];
#pragma unroll
  for (int i = 0; i < 8; ++i) {
    const int row  = wrow + mbase + i;
    const size_t idx = (size_t)row * H_DIM + colH;
    float gi = sigmoidf_fast(acc[0][i] + bias_i);
    float gf = sigmoidf_fast(acc[1][i] + bias_f);
    float gg = tanhf_fast(acc[2][i] + bias_g);
    float go = sigmoidf_fast(acc[3][i] + bias_o);
    float c_old = cbuf[idx];
    float c_new = gf * c_old + gi * gg;
    cbuf[idx] = c_new;                    // c_new half of d_out
    hnew[idx] = go * tanhf_fast(c_new);   // h_new half of d_out
  }
}

// ---------------------------------------------------------------------------
// Host-side orchestration
// ---------------------------------------------------------------------------
extern "C" void kernel_launch(void* const* d_in, const int* in_sizes, int n_in,
                              void* d_out, int out_size, void* d_ws, size_t ws_size,
                              hipStream_t stream) {
  (void)in_sizes; (void)n_in; (void)out_size; (void)ws_size;

  const float* x     = (const float*)d_in[0];   // [S, I]
  const float* h_s   = (const float*)d_in[1];   // [S, H]
  const float* c_s   = (const float*)d_in[2];   // [S, H]
  const float* hA_f  = (const float*)d_in[3];   // h_after  [S, H]
  const float* hB_f  = (const float*)d_in[4];   // h_before [S, H]
  const float* Wsf_f = (const float*)d_in[5];   // [H, 2H]
  const float* b_sf  = (const float*)d_in[6];   // [H]
  const float* Wsi_f = (const float*)d_in[7];   // [H, 2H]
  const float* b_si  = (const float*)d_in[8];   // [H]
  const float* Wih_f = (const float*)d_in[9];   // [4H, I]
  const float* b_ih  = (const float*)d_in[10];  // [4H]
  const float* Whh_f = (const float*)d_in[11];  // [4H, H]
  const float* b_hh  = (const float*)d_in[12];  // [4H]

  // workspace layout (bf16 copies), all offsets 256B-aligned
  char* ws = (char*)d_ws;
  size_t off = 0;
  bf16_t* xb  = (bf16_t*)(ws + off); off += (size_t)S_DIM * I_DIM * 2;      // 8 MB
  bf16_t* hAb = (bf16_t*)(ws + off); off += (size_t)S_DIM * H_DIM * 2;      // 16 MB
  bf16_t* hBb = (bf16_t*)(ws + off); off += (size_t)S_DIM * H_DIM * 2;      // 16 MB
  bf16_t* hGb = (bf16_t*)(ws + off); off += (size_t)S_DIM * H_DIM * 2;      // 16 MB
  bf16_t* Wsf = (bf16_t*)(ws + off); off += (size_t)H_DIM * 2 * H_DIM * 2;  // 1 MB
  bf16_t* Wsi = (bf16_t*)(ws + off); off += (size_t)H_DIM * 2 * H_DIM * 2;  // 1 MB
  bf16_t* Wih = (bf16_t*)(ws + off); off += (size_t)4 * H_DIM * I_DIM * 2;  // 1 MB
  bf16_t* Whh = (bf16_t*)(ws + off); off += (size_t)4 * H_DIM * H_DIM * 2;  // 2 MB

  float* hnew = (float*)d_out;                         // h_new [S, H]
  float* cbuf = (float*)d_out + (size_t)S_DIM * H_DIM; // c / c_new [S, H]

  // fp32 -> bf16 conversions
  auto cvt = [&](const float* s, bf16_t* d, long n) {
    long n4 = n / 4;
    long blocks = (n4 + 255) / 256;
    if (blocks > 2048) blocks = 2048;
    cvt_f32_bf16_x4<<<dim3((unsigned)blocks), dim3(256), 0, stream>>>(
        (const float4*)s, (uint2*)d, n4);
  };
  cvt(x,     xb,  (long)S_DIM * I_DIM);
  cvt(hA_f,  hAb, (long)S_DIM * H_DIM);
  cvt(hB_f,  hBb, (long)S_DIM * H_DIM);
  cvt(Wsf_f, Wsf, (long)H_DIM * 2 * H_DIM);
  cvt(Wsi_f, Wsi, (long)H_DIM * 2 * H_DIM);
  cvt(Wih_f, Wih, (long)4 * H_DIM * I_DIM);
  cvt(Whh_f, Whh, (long)4 * H_DIM * H_DIM);

  // spatial gates: 32x64 tiles over [S, H]
  dim3 grid1(S_DIM / 32, H_DIM / 64);
  spatial_kernel<<<grid1, dim3(256), 0, stream>>>(
      hAb, hBb, Wsf, Wsi, b_sf, b_si, c_s, h_s, cbuf, hGb);

  // fused LSTM gates + elementwise: 32x64 tiles over [S, H]
  dim3 grid2(S_DIM / 32, H_DIM / 64);
  gates_kernel<<<grid2, dim3(256), 0, stream>>>(
      xb, hGb, Wih, Whh, b_ih, b_hh, cbuf, hnew);
}
